// MultiHopAttention_72696616452421
// MI455X (gfx1250) — matrix operand order repaired and checked by reading, not compiled, not run on hardware
//
#include <hip/hip_runtime.h>
#include <hip/hip_bf16.h>
#include <math.h>

// Problem constants (match reference)
#define BB 32
#define SS 2048
#define HH 1024
#define NSCH 8           // S-chunks for column reductions

// Fused GEMM tile config
#define TM 64            // rows (s) per workgroup
#define NBW 64           // N-block width (4 x 16 subtiles per wave)
#define KBW 128          // K-block staged for B
#define FUSED_THREADS 128  // 4 waves (wave32)

// LDS padded strides (in bf16 elements): +8 halfs = +16B so successive rows
// land 4 banks apart (row stride no longer a multiple of the 256B bank period)
#define AST (HH + 8)     // A row stride
#define BST (KBW + 8)    // B row stride

#define AS1 __attribute__((address_space(1)))
#define AS3 __attribute__((address_space(3)))

typedef __attribute__((ext_vector_type(16))) __bf16 bf16x16;
typedef __attribute__((ext_vector_type(8)))  float  f32x8;
typedef int v4i __attribute__((vector_size(16)));   // matches builtin param type

// ---------------------------------------------------------------------------
// W [H][H] f32 (row-major, k-major)  ->  Wt [H(n)][H(k)] bf16 (transposed)
// ---------------------------------------------------------------------------
__global__ void convert_w_t(const float* __restrict__ W, __bf16* __restrict__ Wt) {
    int idx = blockIdx.x * 256 + threadIdx.x;       // over H*H
    int k = idx >> 10;          // row of W
    int n = idx & (HH - 1);     // col of W
    Wt[(size_t)n * HH + k] = (__bf16)W[idx];
}

// ---------------------------------------------------------------------------
// Partial column reduction over an S-chunk:
//   part[sc][b][h] = sum_{s in chunk} x[b,s,h] * (wgt ? wgt[b,s] : 1)
// grid: (H/256, NSCH, B), block 256
// ---------------------------------------------------------------------------
__global__ void colsum_partial(const float* __restrict__ x,
                               const float* __restrict__ wgt,
                               float* __restrict__ part) {
    const int h  = blockIdx.x * 256 + threadIdx.x;
    const int sc = blockIdx.y;
    const int b  = blockIdx.z;
    const int s0 = sc * (SS / NSCH);
    const float* xp = x + ((size_t)b * SS + s0) * HH + h;
    float acc = 0.f;
    if (wgt) {
        const float* wp = wgt + (size_t)b * SS + s0;
#pragma unroll 4
        for (int s = 0; s < SS / NSCH; ++s) {
            if ((s & 31) == 0 && s + 32 < SS / NSCH)
                __builtin_prefetch(xp + (size_t)(s + 32) * HH, 0, 1);
            acc += xp[(size_t)s * HH] * wp[s];
        }
    } else {
#pragma unroll 4
        for (int s = 0; s < SS / NSCH; ++s) {
            if ((s & 31) == 0 && s + 32 < SS / NSCH)
                __builtin_prefetch(xp + (size_t)(s + 32) * HH, 0, 1);
            acc += xp[(size_t)s * HH];
        }
    }
    part[((size_t)sc * BB + b) * HH + h] = acc;
}

// ---------------------------------------------------------------------------
// mean finish: out0 = mean, m0 = mean*q, m1 = 2*mean*q   (grid: H/256 x B)
// ---------------------------------------------------------------------------
__global__ void mean_finish(const float* __restrict__ part,
                            const float* __restrict__ q,
                            float* __restrict__ out0,
                            float* __restrict__ m0,
                            float* __restrict__ m1) {
    const int h = blockIdx.x * 256 + threadIdx.x;
    const int b = blockIdx.y;
    float s = 0.f;
#pragma unroll
    for (int c = 0; c < NSCH; ++c) s += part[((size_t)c * BB + b) * HH + h];
    const float mean = s * (1.0f / SS);
    const float mq = mean * q[(size_t)b * HH + h];
    out0[(size_t)b * HH + h] = mean;
    m0[(size_t)b * HH + h]   = mq;
    m1[(size_t)b * HH + h]   = 2.0f * mq;
}

// att finish: out = sum of partials   (grid: H/256 x B)
__global__ void att_finish(const float* __restrict__ part, float* __restrict__ out) {
    const int h = blockIdx.x * 256 + threadIdx.x;
    const int b = blockIdx.y;
    float s = 0.f;
#pragma unroll
    for (int c = 0; c < NSCH; ++c) s += part[((size_t)c * BB + b) * HH + h];
    out[(size_t)b * HH + h] = s;
}

// ---------------------------------------------------------------------------
// v[b,d] = tanh( sum_h m[b,h]*Wm[h,d] ) * Wh[d]    (grid: H/256 x B)
// ---------------------------------------------------------------------------
__global__ void gate_v(const float* __restrict__ m, const float* __restrict__ Wm,
                       const float* __restrict__ Wh, float* __restrict__ v) {
    const int d = blockIdx.x * 256 + threadIdx.x;
    const int b = blockIdx.y;
    const float* mb = m + (size_t)b * HH;
    float acc = 0.f;
#pragma unroll 4
    for (int h = 0; h < HH; ++h) acc += mb[h] * Wm[(size_t)h * HH + d];
    v[(size_t)b * HH + d] = tanhf(acc) * Wh[d];
}

// ---------------------------------------------------------------------------
// Softmax over S, in place.  grid: B, block: 256
// ---------------------------------------------------------------------------
__global__ void softmax_seq(float* __restrict__ U) {
    const int b = blockIdx.x;
    float* u = U + (size_t)b * SS;
    __shared__ float red[256];
    float mx = -INFINITY;
    for (int s = threadIdx.x; s < SS; s += 256) mx = fmaxf(mx, u[s]);
    red[threadIdx.x] = mx; __syncthreads();
    for (int st = 128; st > 0; st >>= 1) {
        if (threadIdx.x < st) red[threadIdx.x] = fmaxf(red[threadIdx.x], red[threadIdx.x + st]);
        __syncthreads();
    }
    mx = red[0]; __syncthreads();
    float sum = 0.f;
    for (int s = threadIdx.x; s < SS; s += 256) { float e = __expf(u[s] - mx); u[s] = e; sum += e; }
    red[threadIdx.x] = sum; __syncthreads();
    for (int st = 128; st > 0; st >>= 1) {
        if (threadIdx.x < st) red[threadIdx.x] += red[threadIdx.x + st];
        __syncthreads();
    }
    const float inv = 1.0f / red[0];
    for (int s = threadIdx.x; s < SS; s += 256) u[s] *= inv;
}

// ---------------------------------------------------------------------------
// Fused: U[b,s] = sum_d tanh( sum_h x[b,s,h]*W[h,d] ) * v[b,d]
//   x  : [B,S,H] f32 (converted to bf16 into LDS, full-K resident)
//   Wt : [H(n)][H(k)] bf16 (pre-transposed)
// grid: (S/TM, B), block: 128 (4 waves), dynamic LDS = (TM*AST + NBW*BST)*2 B
// ---------------------------------------------------------------------------
union Frag { bf16x16 v; uint4 q[2]; };

__global__ __launch_bounds__(FUSED_THREADS)
void fused_score(const float* __restrict__ x, const __bf16* __restrict__ Wt,
                 const float* __restrict__ v, float* __restrict__ U) {
    extern __shared__ __bf16 lds[];
    __bf16* Alds = lds;                     // [TM][AST]
    __bf16* Blds = lds + (size_t)TM * AST;  // [NBW][BST]

    const int b    = blockIdx.y;
    const int s0   = blockIdx.x * TM;
    const int tid  = threadIdx.x;
    const int lane = tid & 31;
    const int wv   = tid >> 5;        // wave id 0..3, owns rows wv*16..wv*16+15
    const int mcol = lane & 15;       // M-row (A) / N-col (B,C) within 16
    const int hi   = lane >> 4;       // half-wave selector

    // ---- stage A: convert inputs[b, s0..s0+TM) f32 -> bf16 in LDS (read once) ----
    {
        const float4* src = (const float4*)(x + ((size_t)b * SS + s0) * HH);
        for (int i = tid; i < TM * HH / 4; i += FUSED_THREADS) {
            float4 f = src[i];
            const int row = i >> 8;               // (i*4)/HH
            const int k   = (i * 4) & (HH - 1);
            __bf16* dst = Alds + (size_t)row * AST + k;
            dst[0] = (__bf16)f.x; dst[1] = (__bf16)f.y;
            dst[2] = (__bf16)f.z; dst[3] = (__bf16)f.w;
        }
    }

    float uacc[8] = {};
    const float* vb = v + (size_t)b * HH;
    const int arow = wv * 16 + mcol;

    for (int nb = 0; nb < HH; nb += NBW) {
        f32x8 c[4] = {};
        for (int kb = 0; kb < HH; kb += KBW) {
            __syncthreads();   // all waves done reading previous B block (and A staged)
            // stage Blds[n][k] = Wt[(nb+n)*H + kb + k], 64x128 bf16, 16B granules
#if __has_builtin(__builtin_amdgcn_global_load_async_to_lds_b128)
            for (int i = tid; i < NBW * KBW / 8; i += FUSED_THREADS) {
                const int n  = i >> 4;       // KBW/8 = 16 granules per row
                const int kq = i & 15;
                const __bf16* gsrc = Wt + (size_t)(nb + n) * HH + kb + kq * 8;
                __bf16* ldst = Blds + (size_t)n * BST + kq * 8;
                __builtin_amdgcn_global_load_async_to_lds_b128(
                    (AS1 v4i*)gsrc, (AS3 v4i*)ldst, 0, 0);
            }
#if __has_builtin(__builtin_amdgcn_s_wait_asynccnt)
            __builtin_amdgcn_s_wait_asynccnt(0);
#else
            asm volatile("s_wait_asynccnt 0x0" ::: "memory");
#endif
#else
            for (int i = tid; i < NBW * KBW / 8; i += FUSED_THREADS) {
                const int n  = i >> 4;
                const int kq = i & 15;
                *(uint4*)(Blds + (size_t)n * BST + kq * 8) =
                    *(const uint4*)(Wt + (size_t)(nb + n) * HH + kb + kq * 8);
            }
#endif
            __syncthreads();
            for (int ks = 0; ks < KBW; ks += 32) {
                // A fragment (16x32, half-lane K split per ISA layout)
                Frag a;
                const __bf16* ap = Alds + (size_t)arow * AST + kb + ks;
                a.q[0] = *(const uint4*)(ap + hi * 8);
                a.q[1] = *(const uint4*)(ap + 16 + hi * 8);
                // load all 4 B fragments first, then issue the WMMA chain
                Frag bf[4];
#pragma unroll
                for (int sub = 0; sub < 4; ++sub) {
                    const __bf16* bp = Blds + (size_t)(sub * 16 + mcol) * BST + ks + hi * 8;
                    bf[sub].q[0] = *(const uint4*)(bp);
                    bf[sub].q[1] = *(const uint4*)(bp + 16);
                }
#pragma unroll
                for (int sub = 0; sub < 4; ++sub) {
                    c[sub] = __builtin_amdgcn_wmma_f32_16x16x32_bf16(
                        false, a.v, false, bf[sub].v, (short)0, c[sub], false, false);
                }
            }
        }
        // epilogue: tanh, scale by v[b, n], accumulate partial row sums
#pragma unroll
        for (int sub = 0; sub < 4; ++sub) {
            const float vc = vb[nb + sub * 16 + mcol];
#pragma unroll
            for (int r = 0; r < 8; ++r) uacc[r] += tanhf(c[sub][r]) * vc;
        }
    }

    // reduce over the 16 N-columns (lanes) within each half-wave
#pragma unroll
    for (int r = 0; r < 8; ++r) {
        float t = uacc[r];
        t += __shfl_xor(t, 1, 32);
        t += __shfl_xor(t, 2, 32);
        t += __shfl_xor(t, 4, 32);
        t += __shfl_xor(t, 8, 32);
        uacc[r] = t;
    }
    if (mcol == 0) {
        // lane 0: rows M=0..7 ; lane 16: rows M=8..15  (C layout)
        float* up = U + (size_t)b * SS + s0 + wv * 16 + hi * 8;
#pragma unroll
        for (int r = 0; r < 8; ++r) up[r] = uacc[r];
    }
}

// ---------------------------------------------------------------------------
extern "C" void kernel_launch(void* const* d_in, const int* in_sizes, int n_in,
                              void* d_out, int out_size, void* d_ws, size_t ws_size,
                              hipStream_t stream) {
    const float* x   = (const float*)d_in[0];   // [B,S,H]
    const float* q   = (const float*)d_in[1];   // [B,H]
    const float* W1  = (const float*)d_in[2];   // [H,H]
    const float* Wm1 = (const float*)d_in[3];   // [H,H]
    const float* Wh1 = (const float*)d_in[4];   // [H,1]
    const float* W2  = (const float*)d_in[5];
    const float* Wm2 = (const float*)d_in[6];
    const float* Wh2 = (const float*)d_in[7];

    float* out0 = (float*)d_out;                // [B,H]
    float* out1 = out0 + (size_t)BB * HH;
    float* out2 = out1 + (size_t)BB * HH;

    // workspace layout
    float* U    = (float*)d_ws;                 // B*S
    float* part = U + (size_t)BB * SS;          // NSCH*B*H
    float* m0   = part + (size_t)NSCH * BB * HH;
    float* m1   = m0 + (size_t)BB * HH;
    float* vbuf = m1 + (size_t)BB * HH;
    __bf16* Wt1 = (__bf16*)(vbuf + (size_t)BB * HH);
    __bf16* Wt2 = Wt1 + (size_t)HH * HH;

    const dim3 blk256(256);
    const dim3 gHB(HH / 256, BB);
    const dim3 gCol(HH / 256, NSCH, BB);
    const dim3 gGemm(SS / TM, BB);
    const size_t ldsBytes = ((size_t)TM * AST + (size_t)NBW * BST) * sizeof(__bf16); // ~146 KB

    // weight convert+transpose (once)
    convert_w_t<<<HH * HH / 256, blk256, 0, stream>>>(W1, Wt1);
    convert_w_t<<<HH * HH / 256, blk256, 0, stream>>>(W2, Wt2);

    // mean + m0/m1
    colsum_partial<<<gCol, blk256, 0, stream>>>(x, nullptr, part);
    mean_finish<<<gHB, blk256, 0, stream>>>(part, q, out0, m0, m1);

    // hop 1
    gate_v<<<gHB, blk256, 0, stream>>>(m0, Wm1, Wh1, vbuf);
    fused_score<<<gGemm, FUSED_THREADS, ldsBytes, stream>>>(x, Wt1, vbuf, U);
    softmax_seq<<<BB, blk256, 0, stream>>>(U);
    colsum_partial<<<gCol, blk256, 0, stream>>>(x, U, part);
    att_finish<<<gHB, blk256, 0, stream>>>(part, out1);

    // hop 2
    gate_v<<<gHB, blk256, 0, stream>>>(m1, Wm2, Wh2, vbuf);
    fused_score<<<gGemm, FUSED_THREADS, ldsBytes, stream>>>(x, Wt2, vbuf, U);
    softmax_seq<<<BB, blk256, 0, stream>>>(U);
    colsum_partial<<<gCol, blk256, 0, stream>>>(x, U, part);
    att_finish<<<gHB, blk256, 0, stream>>>(part, out2);
}
